// PointTransformerConv1_15693810499810
// MI455X (gfx1250) — matrix operand order repaired and checked by reading, not compiled, not run on hardware
//
#include <hip/hip_runtime.h>
#include <hip/hip_bf16.h>
#include <math.h>

#define NB 8
#define MP 2048
#define KN 64
#define CC 64
#define NP (NB*MP)
#define RAD2 0.04f
#define INVBN 0.9999950000374997f   /* 1/sqrt(1+1e-5) */

typedef _Float16 v16h __attribute__((ext_vector_type(16)));
typedef _Float16 v8h  __attribute__((ext_vector_type(8)));
typedef float    v8f  __attribute__((ext_vector_type(8)));
typedef unsigned int u32x4 __attribute__((ext_vector_type(4)));
typedef int          i32x4 __attribute__((ext_vector_type(4)));
typedef int          i32x8 __attribute__((ext_vector_type(8)));

__device__ __forceinline__ v8f wmma_f16(v16h a, v16h b, v8f c) {
  // D = A(16x32 f16) * B(32x16 f16) + C(16x16 f32)
  return __builtin_amdgcn_wmma_f32_16x16x32_f16(false, a, false, b, (short)0, c,
                                                false, false);
}

#if defined(__has_builtin)
#if __has_builtin(__builtin_amdgcn_tensor_load_to_lds)
#define HAVE_TDM 1
#endif
#endif

#ifdef HAVE_TDM
// Tensor Data Mover: 2D tile load Global -> LDS.
// D# group0: count=1 | lds_addr | global_addr[56:0] | type=2.
// D# group1: data_size, tensor_dim0/1, tile_dim0/1 (tile_dim2=0 -> 2D),
//            tensor_dim0_stride. Groups 2/3 zero (unused for 2D).
__device__ __forceinline__ void tdm_load_2d(unsigned int lds_addr,
                                            const void* gaddr,
                                            unsigned int dim0, unsigned int dim1,
                                            unsigned int tile0, unsigned int tile1,
                                            unsigned int stride0,
                                            unsigned int dsz_log2) {
  unsigned long long ga = (unsigned long long)(size_t)gaddr;
  u32x4 g0;
  g0[0] = 1u;                                   // count=1, user mode
  g0[1] = lds_addr;                             // bits 63:32
  g0[2] = (unsigned int)ga;                     // global_addr[31:0]
  g0[3] = (unsigned int)((ga >> 32) & 0x1FFFFFFu) | (2u << 30); // addr|type=2
  i32x8 g1;
  g1[0] = (int)(dsz_log2 << 16);                // wg_mask=0, data_size
  g1[1] = (int)((dim0 & 0xFFFFu) << 16);        // tensor_dim0[15:0] @ 63:48
  g1[2] = (int)(((dim0 >> 16) & 0xFFFFu) | ((dim1 & 0xFFFFu) << 16));
  g1[3] = (int)(((dim1 >> 16) & 0xFFFFu) | ((tile0 & 0xFFFFu) << 16));
  g1[4] = (int)(tile1 & 0xFFFFu);               // tile_dim1; tile_dim2=0
  g1[5] = (int)stride0;                         // tensor_dim0_stride[31:0]
  g1[6] = 0;
  g1[7] = 0;
  i32x4 z4 = {0, 0, 0, 0};
#if __clang_major__ >= 23
  i32x8 z8 = {0, 0, 0, 0, 0, 0, 0, 0};
  __builtin_amdgcn_tensor_load_to_lds(g0, g1, z4, z4, z8, 0);
#else
  __builtin_amdgcn_tensor_load_to_lds(g0, g1, z4, z4, 0);
#endif
}
#endif

// ---------------------------------------------------------------------------
// Kernel 1: pack weight matrices into WMMA B-operand lane layout (f16).
// B tile (kt,nt) of a [64K x 64N] matrix: lane<16 holds col n=nt*16+lane,
// halves h<8 <-> K=kt*32+h, h>=8 <-> K=kt*32+8+h ; lane>=16 adds 8 to K.
// pack[((kt*4+nt)*32 + lane)*16 + h].
// Matrices: 0=lin_w(v) 1=lin_src_w(k) 2=lin_dst_w(q) 3=attn_w, then
// pos_w padded from K=6 to K=32 (single k-step) at offset 4*4096.
// ---------------------------------------------------------------------------
__global__ __launch_bounds__(256) void pack_kernel(
    const float* __restrict__ lin_w, const float* __restrict__ src_w,
    const float* __restrict__ dst_w, const float* __restrict__ attn_w,
    const float* __restrict__ pos_w, _Float16* __restrict__ pack) {
  int i = blockIdx.x * 256 + threadIdx.x;   // 72 blocks * 256 = 18432 exactly
  if (i < 4 * 4096) {
    int mtx = i >> 12;
    int r = i & 4095;
    int h = r & 15, ln = (r >> 4) & 31, nt = (r >> 9) & 3, kt = r >> 11;
    int K = kt * 32 + (h < 8 ? h : h + 8) + (ln < 16 ? 0 : 8);
    int n = nt * 16 + (ln & 15);
    const float* W = (mtx == 0) ? lin_w : (mtx == 1) ? src_w
                    : (mtx == 2) ? dst_w : attn_w;
    pack[i] = (_Float16)W[K * 64 + n];
  } else {
    int r = i - 4 * 4096;                   // < 2048, kt == 0 only
    int h = r & 15, ln = (r >> 4) & 31, nt = (r >> 9) & 3;
    int K = (h < 8 ? h : h + 8) + (ln < 16 ? 0 : 8);
    int n = nt * 16 + (ln & 15);
    pack[i] = (_Float16)((K < 6) ? pos_w[K * 64 + n] : 0.0f);
  }
}

// ---------------------------------------------------------------------------
// Kernel 2: q/k/v projections. One wave per 16-row tile of x.
// x tile staged to LDS by the Tensor Data Mover (async DMA, TENSORcnt).
// ---------------------------------------------------------------------------
__global__ __launch_bounds__(32) void qkv_kernel(
    const float* __restrict__ x, const _Float16* __restrict__ pack,
    const float* __restrict__ lin_b, const float* __restrict__ src_b,
    const float* __restrict__ dst_b,
    float* __restrict__ vbuf, float* __restrict__ kbuf, float* __restrict__ qbuf) {
  __shared__ __attribute__((aligned(16))) float xt[16 * CC];
  int lane = threadIdx.x;
  int rowbase = blockIdx.x * 16;
#ifdef HAVE_TDM
  // TDM: 2D tile 64x16 f32, row stride 64 elems, into LDS at xt.
  tdm_load_2d((unsigned int)(size_t)xt, x + (size_t)rowbase * CC,
              CC, 16, CC, 16, CC, 2);
  __builtin_amdgcn_s_wait_tensorcnt(0);
#else
  const float4* xs = (const float4*)(x + (size_t)rowbase * CC);
  float4* xd = (float4*)xt;
  for (int i = lane; i < 256; i += 32) xd[i] = xs[i];
#endif
  // single wave: same-wave DS ordering guarantees visibility; no barrier

  int l15 = lane & 15, hi = lane >> 4;
  // A operand (16x32 f16): lane<16 row=l15 halves {K 0..7,16..23}; hi +8
  v16h a0, a1;
  for (int h = 0; h < 16; ++h) {
    int K0 = (h < 8 ? h : h + 8) + hi * 8;
    a0[h] = (_Float16)xt[l15 * CC + K0];
    a1[h] = (_Float16)xt[l15 * CC + 32 + K0];
  }
  for (int m = 0; m < 3; ++m) {
    const _Float16* Bp = pack + m * 4096;
    const float* bias = (m == 0) ? lin_b : (m == 1) ? src_b : dst_b;
    float* ob = (m == 0) ? vbuf : (m == 1) ? kbuf : qbuf;
    for (int nt = 0; nt < 4; ++nt) {
      v16h b0 = *(const v16h*)&Bp[((0 * 4 + nt) * 32 + lane) * 16];
      v16h b1 = *(const v16h*)&Bp[((1 * 4 + nt) * 32 + lane) * 16];
      v8f acc = {0.f, 0.f, 0.f, 0.f, 0.f, 0.f, 0.f, 0.f};
      acc = wmma_f16(a0, b0, acc);
      acc = wmma_f16(a1, b1, acc);
      int c = nt * 16 + l15;
      float bs = bias[c];
      for (int v = 0; v < 8; ++v)
        ob[(size_t)(rowbase + v + hi * 8) * CC + c] = acc[v] + bs;
    }
  }
}

// ---------------------------------------------------------------------------
// Kernel 3: per-point 64-nearest-neighbor selection within its cloud.
// One 256-thread block per point: d2 row in LDS, 64 argmin rounds.
// ---------------------------------------------------------------------------
__global__ __launch_bounds__(256) void nbr_kernel(
    const float* __restrict__ pos, int* __restrict__ idxb,
    float* __restrict__ d2b) {
  __shared__ float d2s[MP];
  __shared__ float wmin[8];
  __shared__ int warg[8];
  int p = blockIdx.x;
  int t = threadIdx.x;
  int base = (p >> 11) << 11;   // cloud start (M=2048)
  float px = pos[p * 3 + 0], py = pos[p * 3 + 1], pz = pos[p * 3 + 2];
  for (int j = t; j < MP; j += 256) {
    float dx = px - pos[(base + j) * 3 + 0];
    float dy = py - pos[(base + j) * 3 + 1];
    float dz = pz - pos[(base + j) * 3 + 2];
    d2s[j] = dx * dx + dy * dy + dz * dz;
  }
  __syncthreads();
  int lane = t & 31, wv = t >> 5;
  for (int round = 0; round < KN; ++round) {
    float best = 3.0e38f;
    int barg = 0;
    for (int j = t; j < MP; j += 256) {
      float d = d2s[j];
      if (d < best) { best = d; barg = j; }
    }
    for (int off = 16; off > 0; off >>= 1) {
      float ob = __shfl_xor(best, off, 32);
      int oa = __shfl_xor(barg, off, 32);
      if (ob < best || (ob == best && oa < barg)) { best = ob; barg = oa; }
    }
    if (lane == 0) { wmin[wv] = best; warg[wv] = barg; }
    __syncthreads();
    if (t == 0) {
      float b0 = wmin[0]; int a0 = warg[0];
      for (int i = 1; i < 8; ++i)
        if (wmin[i] < b0 || (wmin[i] == b0 && warg[i] < a0)) {
          b0 = wmin[i]; a0 = warg[i];
        }
      idxb[(size_t)p * KN + round] = base + a0;
      d2b[(size_t)p * KN + round] = b0;
      d2s[a0] = 3.0e38f;
    }
    __syncthreads();
  }
}

// ---------------------------------------------------------------------------
// Kernel 4: fused delta / alpha / softmax / aggregate. One wave per point,
// 2 waves per block. S and delta tiles staged in LDS in A-operand-friendly
// order (row stride 72 halves => conflict-free 16-lane ds_load_b128).
// ---------------------------------------------------------------------------
__global__ __launch_bounds__(64) void attn_kernel(
    const float* __restrict__ pos, const float* __restrict__ normal,
    const float* __restrict__ qbuf, const float* __restrict__ kbuf,
    const float* __restrict__ vbuf, const int* __restrict__ idxb,
    const float* __restrict__ d2b, const _Float16* __restrict__ pack,
    const float* __restrict__ pos_b, const float* __restrict__ pos_g,
    const float* __restrict__ pos_bt, const float* __restrict__ attn_b,
    const float* __restrict__ attn_g, const float* __restrict__ attn_bt,
    float* __restrict__ outp) {
  __shared__ __attribute__((aligned(16))) _Float16 sS[2][64 * 72];
  __shared__ __attribute__((aligned(16))) _Float16 sD[2][64 * 72];
  __shared__ int sIdx[2][64];
  __shared__ float sD2[2][64];

  int lane = threadIdx.x & 31;
  int w = threadIdx.x >> 5;
  int p = blockIdx.x * 2 + w;
  int l15 = lane & 15, hi = lane >> 4;
  const _Float16* battn = pack + 3 * 4096;
  const _Float16* bpos = pack + 4 * 4096;

  for (int i = lane; i < KN; i += 32) {
    int j = idxb[(size_t)p * KN + i];
    sIdx[w][i] = j;
    sD2[w][i] = d2b[(size_t)p * KN + i];
    // warm L0/L2 for the upcoming k/v row gathers (global_prefetch_b8)
    __builtin_prefetch(&kbuf[(size_t)j * CC], 0, 3);
    __builtin_prefetch(&vbuf[(size_t)j * CC], 0, 3);
  }
  float px = pos[p * 3 + 0], py = pos[p * 3 + 1], pz = pos[p * 3 + 2];
  float nx = normal[p * 3 + 0], ny = normal[p * 3 + 1], nz = normal[p * 3 + 2];

  // ---- Phase A: delta = relu(bn(rel @ pos_w + pos_b)); S = q - k_j + delta
  for (int mt = 0; mt < 4; ++mt) {
    int jrow[8];
    for (int v = 0; v < 8; ++v) jrow[v] = sIdx[w][mt * 16 + v + hi * 8];
    v16h a;
    for (int h = 0; h < 16; ++h) a[h] = (_Float16)0.0f;
    if (!hi) {  // only K=0..5 nonzero -> lives entirely in low-lane halves
      int j = sIdx[w][mt * 16 + l15];
      a[0] = (_Float16)(px - pos[j * 3 + 0]);
      a[1] = (_Float16)(py - pos[j * 3 + 1]);
      a[2] = (_Float16)(pz - pos[j * 3 + 2]);
      a[3] = (_Float16)(nx - normal[j * 3 + 0]);
      a[4] = (_Float16)(ny - normal[j * 3 + 1]);
      a[5] = (_Float16)(nz - normal[j * 3 + 2]);
    }
    for (int nt = 0; nt < 4; ++nt) {
      int c = nt * 16 + l15;
      v16h b = *(const v16h*)&bpos[(nt * 32 + lane) * 16];
      v8f acc = {0.f, 0.f, 0.f, 0.f, 0.f, 0.f, 0.f, 0.f};
      acc = wmma_f16(a, b, acc);
      float g = pos_g[c], bt = pos_bt[c], pb = pos_b[c];
      float qv = qbuf[(size_t)p * CC + c];
      for (int v = 0; v < 8; ++v) {
        int m = mt * 16 + v + hi * 8;
        float dv = fmaxf(g * (acc[v] + pb) * INVBN + bt, 0.0f);
        sD[w][m * 72 + c] = (_Float16)dv;
        float kv = kbuf[(size_t)jrow[v] * CC + c];
        sS[w][m * 72 + c] = (_Float16)(qv - kv + dv);
      }
    }
  }

  // ---- Phase B: alpha = relu(bn(S @ attn_w + attn_b)); masked softmax over
  // neighbors (rows); out[c] = sum_m w[m][c] * (v[j_m][c] + delta[m][c])
  for (int nt = 0; nt < 4; ++nt) {
    int c = nt * 16 + l15;
    // B operands depend only on (ks, nt): load once per nt
    v16h bv0 = *(const v16h*)&battn[((0 * 4 + nt) * 32 + lane) * 16];
    v16h bv1 = *(const v16h*)&battn[((1 * 4 + nt) * 32 + lane) * 16];
    v8f accA[4];
    for (int mt = 0; mt < 4; ++mt) {
      v8f acc = {0.f, 0.f, 0.f, 0.f, 0.f, 0.f, 0.f, 0.f};
      int r = mt * 16 + l15;
      {
        const _Float16* sp = &sS[w][r * 72 + 0 * 32 + hi * 8];
        v8h c0 = *(const v8h*)sp;
        v8h c1 = *(const v8h*)(sp + 16);
        v16h av;
        for (int h = 0; h < 8; ++h) { av[h] = c0[h]; av[8 + h] = c1[h]; }
        acc = wmma_f16(av, bv0, acc);
      }
      {
        const _Float16* sp = &sS[w][r * 72 + 1 * 32 + hi * 8];
        v8h c0 = *(const v8h*)sp;
        v8h c1 = *(const v8h*)(sp + 16);
        v16h av;
        for (int h = 0; h < 8; ++h) { av[h] = c0[h]; av[8 + h] = c1[h]; }
        acc = wmma_f16(av, bv1, acc);
      }
      accA[mt] = acc;
    }
    float g = attn_g[c], bt = attn_bt[c], ab = attn_b[c];
    float mx = -3.0e38f;
    for (int mt = 0; mt < 4; ++mt)
      for (int v = 0; v < 8; ++v) {
        int m = mt * 16 + v + hi * 8;
        float aval = fmaxf(g * (accA[mt][v] + ab) * INVBN + bt, 0.0f);
        float xv = (sD2[w][m] <= RAD2) ? aval : -3.0e38f;
        accA[mt][v] = xv;
        mx = fmaxf(mx, xv);
      }
    mx = fmaxf(mx, __shfl_xor(mx, 16, 32));           // rows 0-7 vs 8-15 halves
    float s = 0.f;
    for (int mt = 0; mt < 4; ++mt)
      for (int v = 0; v < 8; ++v) s += __expf(accA[mt][v] - mx);
    s += __shfl_xor(s, 16, 32);
    float inv = 1.0f / s;
    float o = 0.f;
    for (int mt = 0; mt < 4; ++mt)
      for (int v = 0; v < 8; ++v) {
        int m = mt * 16 + v + hi * 8;
        int j = sIdx[w][m];
        float vv = vbuf[(size_t)j * CC + c];
        float dd = (float)sD[w][m * 72 + c];
        o += __expf(accA[mt][v] - mx) * inv * (vv + dd);
      }
    o += __shfl_xor(o, 16, 32);
    if (!hi) outp[(size_t)p * CC + c] = o;
  }
}

// ---------------------------------------------------------------------------
extern "C" void kernel_launch(void* const* d_in, const int* in_sizes, int n_in,
                              void* d_out, int out_size, void* d_ws,
                              size_t ws_size, hipStream_t stream) {
  (void)in_sizes; (void)n_in; (void)out_size; (void)ws_size;
  const float* x       = (const float*)d_in[0];
  const float* pos     = (const float*)d_in[1];
  const float* normal  = (const float*)d_in[2];
  /* d_in[3] = batch ids: block-contiguous, unused */
  const float* lin_w   = (const float*)d_in[4];
  const float* lin_b   = (const float*)d_in[5];
  const float* src_w   = (const float*)d_in[6];
  const float* src_b   = (const float*)d_in[7];
  const float* dst_w   = (const float*)d_in[8];
  const float* dst_b   = (const float*)d_in[9];
  const float* pos_w   = (const float*)d_in[10];
  const float* pos_b   = (const float*)d_in[11];
  const float* pos_g   = (const float*)d_in[12];
  const float* pos_bt  = (const float*)d_in[13];
  const float* attn_w  = (const float*)d_in[14];
  const float* attn_b  = (const float*)d_in[15];
  const float* attn_g  = (const float*)d_in[16];
  const float* attn_bt = (const float*)d_in[17];
  float* out = (float*)d_out;

  // workspace layout (~20 MB total)
  char* ws = (char*)d_ws;
  _Float16* pack = (_Float16*)ws;                 // 18432 f16, pad to 40960 B
  float* qbuf = (float*)(ws + 40960);             // 16384*64 f32 = 4 MB
  float* kbuf = qbuf + (size_t)NP * CC;           // 4 MB
  float* vbuf = kbuf + (size_t)NP * CC;           // 4 MB
  int* idxb = (int*)(vbuf + (size_t)NP * CC);     // 16384*64 i32 = 4 MB
  float* d2b = (float*)(idxb + (size_t)NP * KN);  // 4 MB

  pack_kernel<<<72, 256, 0, stream>>>(lin_w, src_w, dst_w, attn_w, pos_w, pack);
  qkv_kernel<<<NP / 16, 32, 0, stream>>>(x, pack, lin_b, src_b, dst_b,
                                         vbuf, kbuf, qbuf);
  nbr_kernel<<<NP, 256, 0, stream>>>(pos, idxb, d2b);
  attn_kernel<<<NP / 2, 64, 0, stream>>>(pos, normal, qbuf, kbuf, vbuf, idxb,
                                         d2b, pack, pos_b, pos_g, pos_bt,
                                         attn_b, attn_g, attn_bt, out);
}
